// E_GCL_base_79482664780354
// MI455X (gfx1250) — matrix operand order repaired and checked by reading, not compiled, not run on hardware
//
#include <hip/hip_runtime.h>

typedef __attribute__((ext_vector_type(16))) _Float16 v16h;
typedef __attribute__((ext_vector_type(8)))  _Float16 v8h;
typedef __attribute__((ext_vector_type(4)))  _Float16 v4h;
typedef __attribute__((ext_vector_type(8)))  float    v8f;

#define N_NODES 50000
#define N_EDGES 800000
#define WAVES 4

#define AS1 __attribute__((address_space(1)))
typedef const AS1 _Float16* gwptr;   // explicit global address-space weight pointer

__device__ __forceinline__ float silu_f(float x) {
    return x / (1.0f + __expf(-x));
}

// Defeat LICM (prevents hoisting 100+ B-fragments into spilled VGPRs) while
// keeping the global address space so loads lower to global_load_b128.
__device__ __forceinline__ gwptr launder_gw(const _Float16* p) {
    unsigned long long v = (unsigned long long)p;
    asm volatile("" : "+s"(v));
    return (gwptr)v;
}

// D = A[16 x 32*KB] @ B[32*KB x 128] + bias, using v_wmma_f32_16x16x32_f16.
// A: LDS row-major, row stride RS halves. WT: global, output-major [128][Kpad].
// cacc[nb][v] holds element (row = v + 8*half, col = nb*16 + lane%16).
template<int KB>
__device__ __forceinline__ void gemm16(const _Float16* __restrict__ A, int RS,
                                       gwptr WT, int Kpad,
                                       const float* __restrict__ bias,
                                       float cacc[8][8], int lane)
{
    const int q  = lane & 15;   // A row / B,D column
    const int hf = lane >> 4;
#pragma unroll
    for (int nb = 0; nb < 8; ++nb) {
        v8f c;
        float bv = bias[nb * 16 + q];
#pragma unroll
        for (int v = 0; v < 8; ++v) c[v] = bv;
#pragma unroll
        for (int kb = 0; kb < KB; ++kb) {
            // A frag: lane q holds row q; K = kb*32 + 8*hf + j and +16
            const _Float16* pa = A + q * RS + kb * 32 + 8 * hf;
            v8h alo = *(const v8h*)pa;
            v8h ahi = *(const v8h*)(pa + 16);
            // B frag: lane holds column n = nb*16+q; K = kb*32 + 8*hf + j and +16
            gwptr pb = WT + (nb * 16 + q) * Kpad + kb * 32 + 8 * hf;
            v8h blo = *(const AS1 v8h*)pb;
            v8h bhi = *(const AS1 v8h*)(pb + 16);
            v16h a, b;
#pragma unroll
            for (int j = 0; j < 8; ++j) {
                a[j] = alo[j]; a[8 + j] = ahi[j];
                b[j] = blo[j]; b[8 + j] = bhi[j];
            }
            c = __builtin_amdgcn_wmma_f32_16x16x32_f16(
                    false, a, false, b, (short)0, c, false, false);
        }
#pragma unroll
        for (int v = 0; v < 8; ++v) cacc[nb][v] = c[v];
    }
}

__device__ __forceinline__ v4h cvt4(float4 x) {
    v4h r;
    r[0] = (_Float16)x.x; r[1] = (_Float16)x.y;
    r[2] = (_Float16)x.z; r[3] = (_Float16)x.w;
    return r;
}

// Convert W[K][128] (f32, in-major) -> WT[128][Kpad] (f16, out-major, zero pad)
__global__ void prep_w(const float* __restrict__ w, _Float16* __restrict__ wt,
                       int K, int Kpad)
{
    int total = 128 * Kpad;
    for (int idx = blockIdx.x * blockDim.x + threadIdx.x; idx < total;
         idx += gridDim.x * blockDim.x) {
        int o = idx / Kpad;
        int k = idx - o * Kpad;
        wt[idx] = (k < K) ? (_Float16)w[k * 128 + o] : (_Float16)0.0f;
    }
}

// Fused edge pipeline: ef_in -> edge MLP -> atomic agg, plus acc-edge MLP -> atomic agg_c
__global__ __launch_bounds__(WAVES * 32)
void edge_kernel(const float* __restrict__ h, const int* __restrict__ ei,
                 const float* __restrict__ coord, const float* __restrict__ eattr,
                 const float* __restrict__ prompt,
                 const float* __restrict__ eb1, const float* __restrict__ eb2,
                 const float* __restrict__ aeb1, const float* __restrict__ aew2,
                 const _Float16* __restrict__ ew1T_g, const _Float16* __restrict__ ew2T_g,
                 const _Float16* __restrict__ aw1T_g,
                 float* __restrict__ agg, float* __restrict__ aggc)
{
    __shared__ __align__(16) _Float16 act[WAVES][16][416];
    __shared__ float cdv[WAVES][16][3];
    __shared__ float rnv[WAVES][16];
    __shared__ float sbv[WAVES][16];
    __shared__ int   rowsS[WAVES][16];

    const int lane = threadIdx.x & 31;
    const int wv   = threadIdx.x >> 5;
    const int q    = lane & 15;
    const int hf   = lane >> 4;
    _Float16* A = &act[wv][0][0];

    // tile-invariant: prompt as f16, one float4-chunk per lane
    v4h ph = cvt4(((const float4*)prompt)[lane]);

    const int NT = N_EDGES / 16;
    int wid = blockIdx.x * WAVES + wv;
    int wstride = gridDim.x * WAVES;

    for (int tile = wid; tile < NT; tile += wstride) {
        gwptr ew1T = launder_gw(ew1T_g);
        gwptr ew2T = launder_gw(ew2T_g);
        gwptr aw1T = launder_gw(aw1T_g);

        int e0 = tile * 16;
        // --- geometry (lanes 0..15, one edge each) ---
        if (lane < 16) {
            int r = ei[e0 + lane], c = ei[N_EDGES + e0 + lane];
            rowsS[wv][lane] = r;
            float dx = coord[r * 3 + 0] - coord[c * 3 + 0];
            float dy = coord[r * 3 + 1] - coord[c * 3 + 1];
            float dz = coord[r * 3 + 2] - coord[c * 3 + 2];
            float rad = dx * dx + dy * dy + dz * dz;
            cdv[wv][lane][0] = dx; cdv[wv][lane][1] = dy; cdv[wv][lane][2] = dz;
            rnv[wv][lane] = 1.0f / fmaxf(sqrtf(rad), 1e-12f);
            act[wv][lane][256] = (_Float16)rad;       // radial
        }
        // --- stage ef_in = [h_row | h_col | radial | edge_attr | prompt | pad] ---
        for (int e = 0; e < 16; ++e) {
            int r = ei[e0 + e];
            int c = ei[N_EDGES + e0 + e];
            *(v4h*)&act[wv][e][lane * 4]       = cvt4(((const float4*)(h + (size_t)r * 128))[lane]);
            *(v4h*)&act[wv][e][128 + lane * 4] = cvt4(((const float4*)(h + (size_t)c * 128))[lane]);
            for (int i = lane; i < 16;  i += 32) act[wv][e][257 + i] = (_Float16)eattr[(e0 + e) * 16 + i];
            for (int i = lane; i < 128; i += 32) act[wv][e][273 + i] = (_Float16)prompt[i];
            for (int i = lane; i < 15;  i += 32) act[wv][e][401 + i] = (_Float16)0.0f;
        }
        // --- GEMM1: [16x416] @ e_w1 ---
        float c1[8][8];
        gemm16<13>(A, 416, ew1T, 416, eb1, c1, lane);
#pragma unroll
        for (int nb = 0; nb < 8; ++nb)
#pragma unroll
            for (int v = 0; v < 8; ++v)
                act[wv][v + 8 * hf][nb * 16 + q] = (_Float16)silu_f(c1[nb][v]);
        // --- GEMM2: [16x128] @ e_w2 -> edge_feat; atomic agg ---
        float c2[8][8];
        gemm16<4>(A, 416, ew2T, 128, eb2, c2, lane);
#pragma unroll
        for (int nb = 0; nb < 8; ++nb)
#pragma unroll
            for (int v = 0; v < 8; ++v) {
                float ef = silu_f(c2[nb][v]);
                int r = rowsS[wv][v + 8 * hf];
                atomicAdd(&agg[r * 128 + nb * 16 + q], ef);
                act[wv][v + 8 * hf][nb * 16 + q] = (_Float16)ef;
            }
        for (int e = 0; e < 16; ++e)
            *(v4h*)&act[wv][e][128 + lane * 4] = ph;
        // --- GEMM3: [edge_feat|prompt] [16x256] @ ae_w1 ---
        float c3[8][8];
        gemm16<8>(A, 416, aw1T, 256, aeb1, c3, lane);
        // --- dot with ae_w2 (128x1), butterfly reduce over 16 lanes ---
        float part[8];
#pragma unroll
        for (int v = 0; v < 8; ++v) part[v] = 0.0f;
#pragma unroll
        for (int nb = 0; nb < 8; ++nb) {
            float w2 = aew2[nb * 16 + q];
#pragma unroll
            for (int v = 0; v < 8; ++v) part[v] += silu_f(c3[nb][v]) * w2;
        }
#pragma unroll
        for (int v = 0; v < 8; ++v) {
            float p = part[v];
            for (int off = 1; off < 16; off <<= 1) p += __shfl_xor(p, off, 32);
            if (q == 0) sbv[wv][v + 8 * hf] = p;
        }
        // --- trans + atomic agg_c ---
        for (int idx = lane; idx < 48; idx += 32) {
            int e = idx / 3, d = idx - e * 3;
            float val = cdv[wv][e][d] * rnv[wv][e] * sbv[wv][e];
            atomicAdd(&aggc[rowsS[wv][e] * 3 + d], val);
        }
    }
}

// Fused node pipeline: node MLP (+residual) and acc-point MLP; writes all outputs
__global__ __launch_bounds__(WAVES * 32)
void node_kernel(const float* __restrict__ h, const float* __restrict__ coord,
                 const float* __restrict__ prompt,
                 const float* __restrict__ nb1, const float* __restrict__ nb2,
                 const float* __restrict__ apb1, const float* __restrict__ apw2,
                 const _Float16* __restrict__ nw1T_g, const _Float16* __restrict__ nw2T_g,
                 const _Float16* __restrict__ apw1T_g,
                 const float* __restrict__ agg, const float* __restrict__ aggc,
                 float* __restrict__ out)
{
    __shared__ __align__(16) _Float16 act[WAVES][16][384];
    __shared__ float sbv[WAVES][16];

    const int lane = threadIdx.x & 31;
    const int wv   = threadIdx.x >> 5;
    const int q    = lane & 15;
    const int hf   = lane >> 4;
    _Float16* A = &act[wv][0][0];

    float* outH = out;
    float* outC = out + (size_t)N_NODES * 128;
    float* outA = out + (size_t)N_NODES * 131;

    v4h ph = cvt4(((const float4*)prompt)[lane]);

    const int NT = N_NODES / 16;
    int wid = blockIdx.x * WAVES + wv;
    int wstride = gridDim.x * WAVES;

    for (int tile = wid; tile < NT; tile += wstride) {
        gwptr nw1T  = launder_gw(nw1T_g);
        gwptr nw2T  = launder_gw(nw2T_g);
        gwptr apw1T = launder_gw(apw1T_g);

        int n0 = tile * 16;
        // --- stage node_in = [h | agg | prompt] ---
        for (int e = 0; e < 16; ++e) {
            int nd = n0 + e;
            *(v4h*)&act[wv][e][lane * 4]       = cvt4(((const float4*)(h + (size_t)nd * 128))[lane]);
            *(v4h*)&act[wv][e][128 + lane * 4] = cvt4(((const float4*)(agg + (size_t)nd * 128))[lane]);
            *(v4h*)&act[wv][e][256 + lane * 4] = ph;
        }
        // --- GEMM n1: [16x384] @ n_w1 ---
        float c1[8][8];
        gemm16<12>(A, 384, nw1T, 384, nb1, c1, lane);
#pragma unroll
        for (int nb = 0; nb < 8; ++nb)
#pragma unroll
            for (int v = 0; v < 8; ++v)
                act[wv][v + 8 * hf][nb * 16 + q] = (_Float16)silu_f(c1[nb][v]);
        // --- GEMM n2: [16x128] @ n_w2 + residual -> h_out ---
        float c2[8][8];
        gemm16<4>(A, 384, nw2T, 128, nb2, c2, lane);
#pragma unroll
        for (int nb = 0; nb < 8; ++nb)
#pragma unroll
            for (int v = 0; v < 8; ++v) {
                int nd = n0 + v + 8 * hf;
                int col = nb * 16 + q;
                float ho = h[nd * 128 + col] + silu_f(c2[nb][v]);
                outH[nd * 128 + col] = ho;
                act[wv][v + 8 * hf][col] = (_Float16)ho;
            }
        for (int e = 0; e < 16; ++e)
            *(v4h*)&act[wv][e][128 + lane * 4] = ph;
        // --- GEMM n3: [h_out|prompt] [16x256] @ ap_w1 ---
        float c3[8][8];
        gemm16<8>(A, 384, apw1T, 256, apb1, c3, lane);
        float part[8];
#pragma unroll
        for (int v = 0; v < 8; ++v) part[v] = 0.0f;
#pragma unroll
        for (int nb = 0; nb < 8; ++nb) {
            float w2 = apw2[nb * 16 + q];
#pragma unroll
            for (int v = 0; v < 8; ++v) part[v] += silu_f(c3[nb][v]) * w2;
        }
#pragma unroll
        for (int v = 0; v < 8; ++v) {
            float p = part[v];
            for (int off = 1; off < 16; off <<= 1) p += __shfl_xor(p, off, 32);
            if (q == 0) sbv[wv][v + 8 * hf] = p;
        }
        // --- acc = agg_c * s, plus coord passthrough ---
        for (int idx = lane; idx < 48; idx += 32) {
            int e = idx / 3, d = idx - e * 3;
            int nd = n0 + e;
            outC[nd * 3 + d] = coord[nd * 3 + d];
            outA[nd * 3 + d] = aggc[nd * 3 + d] * sbv[wv][e];
        }
    }
}

extern "C" void kernel_launch(void* const* d_in, const int* in_sizes, int n_in,
                              void* d_out, int out_size, void* d_ws, size_t ws_size,
                              hipStream_t stream) {
    const float* h      = (const float*)d_in[0];
    const int*   ei     = (const int*)d_in[1];
    const float* coord  = (const float*)d_in[2];
    const float* eattr  = (const float*)d_in[3];
    const float* prompt = (const float*)d_in[4];
    const float* e_w1 = (const float*)d_in[5];  const float* e_b1 = (const float*)d_in[6];
    const float* e_w2 = (const float*)d_in[7];  const float* e_b2 = (const float*)d_in[8];
    const float* n_w1 = (const float*)d_in[9];  const float* n_b1 = (const float*)d_in[10];
    const float* n_w2 = (const float*)d_in[11]; const float* n_b2 = (const float*)d_in[12];
    const float* ae_w1 = (const float*)d_in[13]; const float* ae_b1 = (const float*)d_in[14];
    const float* ae_w2 = (const float*)d_in[15];
    const float* ap_w1 = (const float*)d_in[16]; const float* ap_b1 = (const float*)d_in[17];
    const float* ap_w2 = (const float*)d_in[18];

    // workspace layout
    float* agg  = (float*)d_ws;                       // N*128 f32
    float* aggc = agg + (size_t)N_NODES * 128;        // N*3 f32
    _Float16* ew1T  = (_Float16*)(aggc + (size_t)N_NODES * 3);
    _Float16* ew2T  = ew1T + 128 * 416;
    _Float16* aw1T  = ew2T + 128 * 128;
    _Float16* nw1T  = aw1T + 128 * 256;
    _Float16* nw2T  = nw1T + 128 * 384;
    _Float16* apw1T = nw2T + 128 * 128;

    hipMemsetAsync(agg, 0, (size_t)(N_NODES * 128 + N_NODES * 3) * sizeof(float), stream);

    prep_w<<<64, 256, 0, stream>>>(e_w1,  ew1T, 401, 416);
    prep_w<<<64, 256, 0, stream>>>(e_w2,  ew2T, 128, 128);
    prep_w<<<64, 256, 0, stream>>>(ae_w1, aw1T, 256, 256);
    prep_w<<<64, 256, 0, stream>>>(n_w1,  nw1T, 384, 384);
    prep_w<<<64, 256, 0, stream>>>(n_w2,  nw2T, 128, 128);
    prep_w<<<64, 256, 0, stream>>>(ap_w1, apw1T, 256, 256);

    edge_kernel<<<1024, WAVES * 32, 0, stream>>>(h, ei, coord, eattr, prompt,
                                                 e_b1, e_b2, ae_b1, ae_w2,
                                                 ew1T, ew2T, aw1T, agg, aggc);
    node_kernel<<<512, WAVES * 32, 0, stream>>>(h, coord, prompt,
                                                n_b1, n_b2, ap_b1, ap_w2,
                                                nw1T, nw2T, apw1T, agg, aggc,
                                                (float*)d_out);
}
